// KGEncoder_14027363188782
// MI455X (gfx1250) — matrix operand order repaired and checked by reading, not compiled, not run on hardware
//
#include <hip/hip_runtime.h>
#include <math.h>

// ---------------------------------------------------------------------------
// RGCN + highway encoder for MI455X (gfx1250, wave32, WMMA).
// Rewrites:
//   sup[...,0:100]   = adj_r @ x            -> bf16 WMMA 16x16x32
//   sup[...,100:200] = rowsum(adj_r) * rel_r-> rank-1 term, folded into bias
//   basis: W_eff = Wb@Ww precomputed; (adj@x)@Wa = adj@(x@Wa)
// adj (binary) converted once to a zero-padded bf16 copy (~181 MB, L2-resident)
// ---------------------------------------------------------------------------

namespace {
constexpr int B_   = 4;
constexpr int R_   = 10;
constexpr int E_   = 1500;
constexpr int EP_  = 1504;            // E padded to 32
constexpr int HID_ = 100;
constexpr int HP_  = 112;             // HID padded to 16
constexpr int KP_  = 128;             // inner-K padded to 32 for x@Wa
constexpr int L_   = 5;
constexpr int NB_  = 3;
constexpr int DIN_ = 200;
constexpr int MT_  = EP_ / 16;        // 94 row tiles
constexpr int NT_  = HP_ / 16;        // 7 col tiles
constexpr int KSB_ = EP_ / 32;        // 47 K-steps per relation in big GEMM
}

typedef __attribute__((ext_vector_type(16))) __bf16       v16bf;
typedef __attribute__((ext_vector_type(8)))  float        v8f;
typedef __attribute__((ext_vector_type(2)))  float        v2f;
typedef __attribute__((ext_vector_type(4)))  unsigned int u32x4;

// Two 16-byte global loads -> one 32-byte WMMA operand register block.
__device__ __forceinline__ void load_op(v16bf& dst, const __bf16* p0, const __bf16* p1) {
    u32x4* d = reinterpret_cast<u32x4*>(&dst);
    d[0] = *reinterpret_cast<const u32x4*>(p0);
    d[1] = *reinterpret_cast<const u32x4*>(p1);
}

// ---------------------------------------------------------------------------
// Prep kernels (tiny)
// ---------------------------------------------------------------------------

// WaT[l][r][n][k] = (Wb[l] @ Ww[l])[r*DIN + k][n], bf16, zero padded, K-contiguous.
__global__ void prep_wa(const float* __restrict__ Wb, const float* __restrict__ Ww,
                        __bf16* __restrict__ WaT) {
    int idx = blockIdx.x * blockDim.x + threadIdx.x;
    if (idx >= L_ * R_ * HP_ * KP_) return;
    int k = idx % KP_;
    int n = (idx / KP_) % HP_;
    int r = (idx / (KP_ * HP_)) % R_;
    int l = idx / (KP_ * HP_ * R_);
    float v = 0.0f;
    if (k < HID_ && n < HID_) {
        int row = r * DIN_ + k;
        for (int nb = 0; nb < NB_; ++nb)
            v += Wb[((size_t)l * (R_ * DIN_) + row) * NB_ + nb] *
                 Ww[((size_t)l * NB_ + nb) * HID_ + n];
    }
    WaT[idx] = (__bf16)v;
}

// Wrel[l][r][n] = sum_d rel[r][d] * W_eff[l][r*DIN+100+d][n]
__global__ void prep_wrel(const float* __restrict__ Wb, const float* __restrict__ Ww,
                          const float* __restrict__ rel, float* __restrict__ Wrel) {
    int idx = blockIdx.x * blockDim.x + threadIdx.x;
    if (idx >= L_ * R_ * HP_) return;
    int n = idx % HP_;
    int r = (idx / HP_) % R_;
    int l = idx / (HP_ * R_);
    float v = 0.0f;
    if (n < HID_) {
        for (int d = 0; d < HID_; ++d) {
            int row = r * DIN_ + HID_ + d;
            float we = 0.0f;
            for (int nb = 0; nb < NB_; ++nb)
                we += Wb[((size_t)l * (R_ * DIN_) + row) * NB_ + nb] *
                      Ww[((size_t)l * NB_ + nb) * HID_ + n];
            v += rel[r * HID_ + d] * we;
        }
    }
    Wrel[idx] = v;
}

// One pass over adj: rowsums S[b][r][i]; optionally a zero-padded bf16 copy.
template <bool WRITE_BF16>
__global__ void prep_adj(const float* __restrict__ adj, float* __restrict__ S,
                         __bf16* __restrict__ adjb) {
    const int i  = blockIdx.x;          // 0..EP-1
    const int br = blockIdx.y;          // 0..B*R-1
    const int t  = threadIdx.x;         // 128 threads
    const float* arow = adj + ((size_t)br * E_ + i) * (size_t)E_;
    __bf16* orow = WRITE_BF16 ? adjb + ((size_t)br * EP_ + i) * EP_ : nullptr;
    float part = 0.0f;
    for (int j = t; j < EP_; j += 128) {
        float v = (i < E_ && j < E_) ? arow[j] : 0.0f;
        part += v;
        if (WRITE_BF16) orow[j] = (__bf16)v;
    }
    __shared__ float red[128];
    red[t] = part;
    __syncthreads();
    for (int s = 64; s > 0; s >>= 1) {
        if (t < s) red[t] += red[t + s];
        __syncthreads();
    }
    if (t == 0) S[(size_t)br * EP_ + i] = red[0];
}

// x[b][i][n] = ent[i][n] broadcast; zero padded.
__global__ void init_x(const float* __restrict__ ent, float* __restrict__ x) {
    int idx = blockIdx.x * blockDim.x + threadIdx.x;
    if (idx >= B_ * EP_ * HP_) return;
    int n = idx % HP_;
    int i = (idx / HP_) % EP_;
    x[idx] = (i < E_ && n < HID_) ? ent[(size_t)i * HID_ + n] : 0.0f;
}

// ---------------------------------------------------------------------------
// Per-layer kernels
// ---------------------------------------------------------------------------

// xb[b][i][k] = bf16(x[b][i][k]) for k<100, i<1500 else 0 (K padded to 128).
__global__ void convert_x(const float* __restrict__ x, __bf16* __restrict__ xb) {
    int idx = blockIdx.x * blockDim.x + threadIdx.x;
    if (idx >= B_ * EP_ * KP_) return;
    int k = idx % KP_;
    int i = (idx / KP_) % EP_;
    int b = idx / (KP_ * EP_);
    float v = (k < HID_ && i < E_) ? x[((size_t)b * EP_ + i) * HP_ + k] : 0.0f;
    xb[idx] = (__bf16)v;
}

// ZT[b][r][n][j] = (x[b] @ Wa[l][r])[j][n]   (stored N-major for B-operand loads)
__global__ void gemm_z(const __bf16* __restrict__ xb, const __bf16* __restrict__ WaT,
                       __bf16* __restrict__ ZT, int layer) {
    const int mt = blockIdx.x;              // 0..93  (rows j)
    const int nt = blockIdx.y;              // 0..6   (cols n)
    const int br = blockIdx.z;              // 0..39
    const int b = br / R_, r = br % R_;
    const int lane = threadIdx.x, lm = lane & 15, lh = lane >> 4;

    v8f acc = {0.f, 0.f, 0.f, 0.f, 0.f, 0.f, 0.f, 0.f};
    const __bf16* ar = xb + ((size_t)b * EP_ + mt * 16 + lm) * KP_;
    const __bf16* wr = WaT + (((size_t)(layer * R_ + r)) * HP_ + nt * 16 + lm) * KP_;

    for (int ks = 0; ks < KP_ / 32; ++ks) {
        v16bf A, Bm;
        const __bf16* pa = ar + ks * 32 + lh * 8;
        load_op(A, pa, pa + 16);                 // A: k = base + {lh*8..+7, 16+lh*8..+7}
        const __bf16* pb = wr + ks * 32 + lh * 8;
        load_op(Bm, pb, pb + 16);                // B: k = base + {lh*8..+7, 16+lh*8..+7}
        acc = __builtin_amdgcn_wmma_f32_16x16x32_bf16(false, A, false, Bm,
                                                      (short)0, acc, false, false);
    }
    __bf16* zt = ZT + ((size_t)(b * R_ + r)) * HP_ * EP_;
    for (int c = 0; c < 8; ++c) {
        int j = mt * 16 + c + lh * 8;            // D row
        int n = nt * 16 + lm;                    // D col
        zt[(size_t)n * EP_ + j] = (__bf16)acc[c];
    }
}

// h1[b][i][n] = bias[l][n] + sum_r S[b][r][i] * Wrel[l][r][n]
__global__ void init_h1(const float* __restrict__ S, const float* __restrict__ Wrel,
                        const float* __restrict__ bias, float* __restrict__ h1, int layer) {
    int idx = blockIdx.x * blockDim.x + threadIdx.x;
    if (idx >= B_ * EP_ * HP_) return;
    int n = idx % HP_;
    int i = (idx / HP_) % EP_;
    int b = idx / (HP_ * EP_);
    float v = 0.0f;
    if (n < HID_) {
        v = bias[layer * HID_ + n];
        for (int r = 0; r < R_; ++r)
            v += S[((size_t)(b * R_ + r)) * EP_ + i] * Wrel[((size_t)(layer * R_ + r)) * HP_ + n];
    }
    h1[idx] = v;
}

// h1[b] += sum_r adj[b][r] @ Z[b][r]; one wave per (16-row strip, b, r), split-K
// over relations with fp32 atomics. ADJ_BF16 selects L2-resident bf16 adj copy.
template <bool ADJ_BF16>
__global__ void gemm_big(const __bf16* __restrict__ adjb, const float* __restrict__ adjf,
                         const __bf16* __restrict__ ZT, float* __restrict__ h1) {
    const int mt = blockIdx.x;              // 0..93
    const int b  = blockIdx.y;              // 0..3
    const int r  = blockIdx.z;              // 0..9
    const int lane = threadIdx.x, lm = lane & 15, lh = lane >> 4;

    v8f acc[NT_];
    for (int n = 0; n < NT_; ++n) acc[n] = (v8f){0.f, 0.f, 0.f, 0.f, 0.f, 0.f, 0.f, 0.f};

    const int row = mt * 16 + lm;
    const __bf16* arow_b = ADJ_BF16 ? adjb + (((size_t)(b * R_ + r)) * EP_ + row) * EP_ : nullptr;
    const float*  arow_f = adjf + (((size_t)(b * R_ + r)) * E_ + row) * (size_t)E_;
    const __bf16* zb     = ZT + ((size_t)(b * R_ + r)) * HP_ * EP_;
    const bool rowok = row < E_;

    for (int ks = 0; ks < KSB_; ++ks) {
        const int jb = ks * 32;
        v16bf A;
        if (ADJ_BF16) {
            const __bf16* pa = arow_b + jb + lh * 8;
            load_op(A, pa, pa + 16);
        } else {
            // guarded fp32 load + in-register convert (fallback if ws too small)
            for (int t = 0; t < 16; ++t) {
                int k = (t < 8) ? (lh * 8 + t) : (16 + lh * 8 + (t - 8));
                int j = jb + k;
                float f = (rowok && j < E_) ? arow_f[j] : 0.0f;
                A[t] = (__bf16)f;
            }
        }
        for (int n = 0; n < NT_; ++n) {
            v16bf Bm;
            const __bf16* pb = zb + (size_t)(n * 16 + lm) * EP_ + jb + lh * 8;
            load_op(Bm, pb, pb + 16);
            acc[n] = __builtin_amdgcn_wmma_f32_16x16x32_bf16(false, A, false, Bm,
                                                             (short)0, acc[n], false, false);
        }
    }
    float* hb = h1 + ((size_t)b * EP_ + mt * 16) * HP_;
    for (int n = 0; n < NT_; ++n)
        for (int c = 0; c < 8; ++c)
            atomicAdd(hb + (size_t)(c + lh * 8) * HP_ + (n * 16 + lm), acc[n][c]);
}

__global__ void relu_k(float* __restrict__ h1) {
    int idx = blockIdx.x * blockDim.x + threadIdx.x;
    if (idx >= B_ * EP_ * HP_) return;
    h1[idx] = fmaxf(h1[idx], 0.0f);
}

// gate = sigmoid(h @ Wh + bh); x = gate*h + (1-gate)*x. fp32 WMMA (K=100=25x4).
__global__ void highway(const float* __restrict__ h, const float* __restrict__ Wh,
                        const float* __restrict__ bh, float* __restrict__ x, int layer) {
    const int mt = blockIdx.x;              // 0..93
    const int nt = blockIdx.y;              // 0..6
    const int b  = blockIdx.z;              // 0..3
    const int lane = threadIdx.x, lm = lane & 15, lh = lane >> 4;

    v8f acc = {0.f, 0.f, 0.f, 0.f, 0.f, 0.f, 0.f, 0.f};
    const float* hb = h + (size_t)b * EP_ * HP_;
    const float* W  = Wh + (size_t)layer * HID_ * HID_;
    const int m = mt * 16 + lm;             // A row
    const int n = nt * 16 + lm;             // B col
    const bool nok = n < HID_;

    for (int ks = 0; ks < HID_ / 4; ++ks) {
        const int k0 = ks * 4 + lh * 2;     // A: k = lh*2 + v ; B: same K striping
        v2f a, bm;
        a[0]  = hb[(size_t)m * HP_ + k0];
        a[1]  = hb[(size_t)m * HP_ + k0 + 1];
        bm[0] = nok ? W[(size_t)k0 * HID_ + n] : 0.0f;
        bm[1] = nok ? W[(size_t)(k0 + 1) * HID_ + n] : 0.0f;
        acc = __builtin_amdgcn_wmma_f32_16x16x4_f32(false, a, false, bm,
                                                    (short)0, acc, false, false);
    }
    const float bhv = nok ? bh[layer * HID_ + n] : 0.0f;
    for (int c = 0; c < 8; ++c) {
        int row = mt * 16 + c + lh * 8;
        float hv = hb[(size_t)row * HP_ + n];
        float g  = 1.0f / (1.0f + expf(-(acc[c] + bhv)));
        size_t xi = ((size_t)b * EP_ + row) * HP_ + n;
        x[xi] = g * hv + (1.0f - g) * x[xi];
    }
}

// out[b][n] = sum_i mask*x / max(count,1)
__global__ void finalize(const float* __restrict__ x, const int* __restrict__ mask,
                         float* __restrict__ out) {
    const int b = blockIdx.x, t = threadIdx.x;  // 128 threads
    __shared__ float red[128];
    __shared__ float denom;
    float p = 0.0f;
    for (int i = t; i < E_; i += 128) p += (float)mask[b * E_ + i];
    red[t] = p;
    __syncthreads();
    for (int s = 64; s > 0; s >>= 1) {
        if (t < s) red[t] += red[t + s];
        __syncthreads();
    }
    if (t == 0) denom = (red[0] == 0.0f) ? 1.0f : red[0];
    __syncthreads();
    if (t < HID_) {
        float s = 0.0f;
        for (int i = 0; i < E_; ++i)
            s += (float)mask[b * E_ + i] * x[((size_t)b * EP_ + i) * HP_ + t];
        out[b * HID_ + t] = s / denom;
    }
}

// ---------------------------------------------------------------------------

extern "C" void kernel_launch(void* const* d_in, const int* in_sizes, int n_in,
                              void* d_out, int out_size, void* d_ws, size_t ws_size,
                              hipStream_t stream) {
    (void)in_sizes; (void)n_in; (void)out_size;
    const float* adj  = (const float*)d_in[0];
    const int*   mask = (const int*)  d_in[1];
    const float* ent  = (const float*)d_in[2];
    const float* rel  = (const float*)d_in[3];
    const float* Wb   = (const float*)d_in[4];
    const float* Ww   = (const float*)d_in[5];
    const float* bias = (const float*)d_in[6];
    const float* Wh   = (const float*)d_in[7];
    const float* bh   = (const float*)d_in[8];
    float* out = (float*)d_out;

    char* ws = (char*)d_ws;
    size_t off = 0;
    auto take = [&](size_t bytes) {
        size_t o = off;
        off += (bytes + 255) & ~(size_t)255;
        return o;
    };
    float*  S    = (float*) (ws + take((size_t)B_ * R_ * EP_ * 4));
    __bf16* WaT  = (__bf16*)(ws + take((size_t)L_ * R_ * HP_ * KP_ * 2));
    float*  Wrel = (float*) (ws + take((size_t)L_ * R_ * HP_ * 4));
    float*  x    = (float*) (ws + take((size_t)B_ * EP_ * HP_ * 4));
    __bf16* xb   = (__bf16*)(ws + take((size_t)B_ * EP_ * KP_ * 2));
    float*  h1   = (float*) (ws + take((size_t)B_ * EP_ * HP_ * 4));
    __bf16* ZT   = (__bf16*)(ws + take((size_t)B_ * R_ * HP_ * EP_ * 2));
    const size_t adjb_bytes = (size_t)B_ * R_ * EP_ * EP_ * 2;
    const bool use_adjb = (off + adjb_bytes) <= ws_size;
    __bf16* adjb = (__bf16*)(ws + off);

    // ---- one-time prep ----
    {
        int n = L_ * R_ * HP_ * KP_;
        prep_wa<<<(n + 255) / 256, 256, 0, stream>>>(Wb, Ww, WaT);
    }
    {
        int n = L_ * R_ * HP_;
        prep_wrel<<<(n + 255) / 256, 256, 0, stream>>>(Wb, Ww, rel, Wrel);
    }
    {
        dim3 g(EP_, B_ * R_);
        if (use_adjb) prep_adj<true><<<g, 128, 0, stream>>>(adj, S, adjb);
        else          prep_adj<false><<<g, 128, 0, stream>>>(adj, S, nullptr);
    }
    {
        int n = B_ * EP_ * HP_;
        init_x<<<(n + 255) / 256, 256, 0, stream>>>(ent, x);
    }

    // ---- 5 RGCN layers ----
    for (int l = 0; l < L_; ++l) {
        {
            int n = B_ * EP_ * KP_;
            convert_x<<<(n + 255) / 256, 256, 0, stream>>>(x, xb);
        }
        {
            dim3 g(MT_, NT_, B_ * R_);
            gemm_z<<<g, 32, 0, stream>>>(xb, WaT, ZT, l);
        }
        {
            int n = B_ * EP_ * HP_;
            init_h1<<<(n + 255) / 256, 256, 0, stream>>>(S, Wrel, bias, h1, l);
        }
        {
            dim3 g(MT_, B_, R_);
            if (use_adjb) gemm_big<true><<<g, 32, 0, stream>>>(adjb, adj, ZT, h1);
            else          gemm_big<false><<<g, 32, 0, stream>>>(nullptr, adj, ZT, h1);
        }
        {
            int n = B_ * EP_ * HP_;
            relu_k<<<(n + 255) / 256, 256, 0, stream>>>(h1);
        }
        {
            dim3 g(MT_, NT_, B_);
            highway<<<g, 32, 0, stream>>>(h1, Wh, bh, x, l);
        }
    }

    finalize<<<B_, 128, 0, stream>>>(x, mask, out);
}
